// IndRNNv2_1460288880807
// MI455X (gfx1250) — compile-verified, matrix operand order checked by reading, not run on hardware
//
#include <hip/hip_runtime.h>

typedef __attribute__((ext_vector_type(16))) __bf16 v16bf;
typedef __attribute__((ext_vector_type(8)))  __bf16 v8bf;
typedef __attribute__((ext_vector_type(8)))  float  v8f;
typedef __attribute__((ext_vector_type(4)))  unsigned v4u;
typedef __attribute__((ext_vector_type(8)))  int      v8i;
typedef __attribute__((ext_vector_type(4)))  int      v4i;

constexpr int kSeq   = 256;
constexpr int kBatch = 32;
constexpr int kIn    = 1024;
constexpr int kHid   = 1024;
constexpr int kM     = kSeq * kBatch;   // 8192 GEMM rows
// GEMM: C[8192,1024] = A[8192,1024] * W[1024,1024]^T (both K-contiguous)

// ---------------------------------------------------------------------------
// Tensor Data Mover: load one 2D tile (128 rows x 16 dwords) global -> LDS,
// padding 4 dwords after every 16 dwords => 80B LDS row stride (bank-optimal).
// D# packing per CDNA5 ISA  (group0 / group1; g2,g3 = 0).
// amdgpu-toolchain (clang-23) 6-arg builtin form.
// ---------------------------------------------------------------------------
__device__ __forceinline__ void tdm_load_tile(unsigned lds_off, const void* gsrc) {
  unsigned long long ga = (unsigned long long)(uintptr_t)gsrc;
  v4u g0;
  g0[0] = 1u;                                   // count=1, user mode
  g0[1] = lds_off;                              // lds_addr (bytes)
  g0[2] = (unsigned)(ga & 0xFFFFFFFFu);         // global_addr[31:0]
  g0[3] = (unsigned)((ga >> 32) & 0x01FFFFFFu)  // global_addr[56:32]
        | (2u << 30);                           // type = 2 ("image")
  v8i g1;
  g1[0] = (int)((2u << 16)     // data_size = 4 bytes
              | (1u << 20)     // pad_enable
              | (3u << 22)     // pad_interval: 16 dwords (64B)
              | (3u << 25));   // pad_amount:   4 dwords (16B)
  g1[1] = (int)(512u << 16);   // tensor_dim0 = 512 dwords (lo16 -> [31:16])
  g1[2] = (int)(8192u << 16);  // tensor_dim0 hi=0 | tensor_dim1 lo = 8192 rows
  g1[3] = (int)(16u << 16);    // tensor_dim1 hi=0 | tile_dim0 = 16 dwords
  g1[4] = 128;                 // tile_dim1 = 128 rows, tile_dim2 = 0 (2D)
  g1[5] = 512;                 // tensor_dim0_stride = 512 dwords
  g1[6] = 0;                   // stride hi / tensor_dim1_stride lo
  g1[7] = 0;
  v4i z4 = {0, 0, 0, 0};
  v8i z8 = {0, 0, 0, 0, 0, 0, 0, 0};
  __builtin_amdgcn_tensor_load_to_lds(g0, g1, z4, z4, z8, 0);
}

__device__ __forceinline__ void wait_tensor0() {
#if __has_builtin(__builtin_amdgcn_s_wait_tensorcnt)
  __builtin_amdgcn_s_wait_tensorcnt(0);
#else
  asm volatile("s_wait_tensorcnt 0x0" ::: "memory");
#endif
}

// ---------------------------------------------------------------------------
// fp32 -> (bf16 hi, bf16 lo) split for ~fp32-accurate bf16x3 WMMA GEMM
// ---------------------------------------------------------------------------
__global__ void split_bf16_kernel(const float* __restrict__ in,
                                  __bf16* __restrict__ hi,
                                  __bf16* __restrict__ lo, int n) {
  int i = blockIdx.x * blockDim.x + threadIdx.x;
  if (i < n) {
    float x = in[i];
    __bf16 h = (__bf16)x;           // RTE to bf16
    hi[i] = h;
    lo[i] = (__bf16)(x - (float)h); // residual
  }
}

// ---------------------------------------------------------------------------
// bf16x3 split GEMM, TDM-staged LDS, double-buffered.
// Block: 128 threads (4 waves, 2x2), block tile 128x128, wave tile 64x64.
// LDS buffer: Ahi|Alo|Bhi|Blo, each 128 rows x 80B = 10240B -> 40KB/buffer.
// ---------------------------------------------------------------------------
#define ARR_BYTES  10240
#define SBUF_BYTES 40960

__device__ __forceinline__ void tdm_fill(char* sbuf, int wave,
    const __bf16* __restrict__ Ahi, const __bf16* __restrict__ Alo,
    const __bf16* __restrict__ Bhi, const __bf16* __restrict__ Blo,
    int M0, int N0, int k0) {
  const __bf16* src;
  unsigned loff;
  if (wave == 0)      { src = Ahi + (size_t)M0 * kIn + k0; loff = 0 * ARR_BYTES; }
  else if (wave == 1) { src = Alo + (size_t)M0 * kIn + k0; loff = 1 * ARR_BYTES; }
  else if (wave == 2) { src = Bhi + (size_t)N0 * kIn + k0; loff = 2 * ARR_BYTES; }
  else                { src = Blo + (size_t)N0 * kIn + k0; loff = 3 * ARR_BYTES; }
  unsigned lds_abs = (unsigned)(uintptr_t)sbuf + loff; // low 32b of flat = LDS addr
  tdm_load_tile(lds_abs, src);
}

// fragment read from padded row-major LDS tile (row stride 80B)
__device__ __forceinline__ v16bf lds_frag(const char* p, int hl) {
  v8bf a = *(const v8bf*)(p + hl * 16);        // K chunk {hl*8 .. +7}
  v8bf b = *(const v8bf*)(p + 32 + hl * 16);   // K chunk {16+hl*8 .. +7}
  v16bf r;
#pragma unroll
  for (int i = 0; i < 8; ++i) { r[i] = a[i]; r[i + 8] = b[i]; }
  return r;
}

__global__ __launch_bounds__(128, 1) void gemm_bf16x3_kernel(
    const __bf16* __restrict__ Ahi, const __bf16* __restrict__ Alo,
    const __bf16* __restrict__ Bhi, const __bf16* __restrict__ Blo,
    const float* __restrict__ bias, float* __restrict__ C) {
  const int K = 1024, N = 1024;
  const int t    = threadIdx.x;
  const int lane = t & 31;
  const int wave = t >> 5;       // 0..3
  const int hl   = lane >> 4;
  const int r    = lane & 15;
  const int wm   = wave >> 1;    // 0..1
  const int wn   = wave & 1;     // 0..1
  const int M0 = blockIdx.x * 128;
  const int N0 = blockIdx.y * 128;

  __shared__ __align__(128) char smem[2 * SBUF_BYTES];   // 80 KB

  v8f acc[4][4];
#pragma unroll
  for (int j = 0; j < 4; ++j) {
    float bv = bias[N0 + wn * 64 + j * 16 + r];
#pragma unroll
    for (int i = 0; i < 4; ++i)
#pragma unroll
      for (int v = 0; v < 8; ++v) acc[i][j][v] = bv;
  }

  // prologue: stage K-step 0 into buffer 0 (4 TDM ops, one per wave)
  tdm_fill(smem, wave, Ahi, Alo, Bhi, Blo, M0, N0, 0);
  wait_tensor0();
  __syncthreads();

  for (int ks = 0; ks < K / 32; ++ks) {
    char* cur = smem + (ks & 1) * SBUF_BYTES;
    if (ks + 1 < K / 32)
      tdm_fill(smem + ((ks + 1) & 1) * SBUF_BYTES, wave,
               Ahi, Alo, Bhi, Blo, M0, N0, (ks + 1) * 32);

    v16bf ah[4], al[4], bh[4], bl[4];
#pragma unroll
    for (int i = 0; i < 4; ++i) {
      const char* p = cur + (wm * 64 + i * 16 + r) * 80;
      ah[i] = lds_frag(p + 0 * ARR_BYTES, hl);
      al[i] = lds_frag(p + 1 * ARR_BYTES, hl);
    }
#pragma unroll
    for (int j = 0; j < 4; ++j) {
      const char* p = cur + (wn * 64 + j * 16 + r) * 80;
      bh[j] = lds_frag(p + 2 * ARR_BYTES, hl);
      bl[j] = lds_frag(p + 3 * ARR_BYTES, hl);
    }
#pragma unroll
    for (int i = 0; i < 4; ++i)
#pragma unroll
      for (int j = 0; j < 4; ++j) {
        acc[i][j] = __builtin_amdgcn_wmma_f32_16x16x32_bf16(
            false, ah[i], false, bh[j], (short)0, acc[i][j], false, false);
        acc[i][j] = __builtin_amdgcn_wmma_f32_16x16x32_bf16(
            false, ah[i], false, bl[j], (short)0, acc[i][j], false, false);
        acc[i][j] = __builtin_amdgcn_wmma_f32_16x16x32_bf16(
            false, al[i], false, bh[j], (short)0, acc[i][j], false, false);
      }

    wait_tensor0();      // next buffer staged (overlapped with compute)
    __syncthreads();     // all waves done reading cur before it is refilled
  }

#pragma unroll
  for (int i = 0; i < 4; ++i) {
    int mb = M0 + wm * 64 + i * 16 + hl * 8;
#pragma unroll
    for (int j = 0; j < 4; ++j) {
      int n = N0 + wn * 64 + j * 16 + r;
#pragma unroll
      for (int v = 0; v < 8; ++v)
        C[(size_t)(mb + v) * N + n] = acc[i][j][v];
    }
  }
}

// ---------------------------------------------------------------------------
// IndRNN recurrence: one thread per (b,h) channel, sequential over S.
// ---------------------------------------------------------------------------
__global__ void indrnn_scan0_kernel(const float* __restrict__ lin,
                                    const float* __restrict__ rec,
                                    __bf16* __restrict__ yhi,
                                    __bf16* __restrict__ ylo,
                                    float* __restrict__ hT) {
  int tid = blockIdx.x * blockDim.x + threadIdx.x;  // 0..32767
  int h = tid & (kHid - 1);
  int b = tid >> 10;
  float rr = rec[h];
  float s = 0.f;
  for (int step = 0; step < kSeq; ++step) {
    size_t idx = ((size_t)step * kBatch + b) * kHid + h;
    float v = lin[idx] + rr * s;
    s = fmaxf(v, 0.f);
    __bf16 hi = (__bf16)s;
    yhi[idx] = hi;
    ylo[idx] = (__bf16)(s - (float)hi);
  }
  hT[b * (2 * kHid) + h] = s;            // hiddens[:, 0:H]
}

__global__ void indrnn_scan1_kernel(const float* __restrict__ lin,
                                    const float* __restrict__ rec,
                                    float* __restrict__ y,
                                    float* __restrict__ hT) {
  int tid = blockIdx.x * blockDim.x + threadIdx.x;
  int h = tid & (kHid - 1);
  int b = tid >> 10;
  float rr = rec[h];
  float s = 0.f;
  for (int step = 0; step < kSeq; ++step) {
    size_t idx = ((size_t)step * kBatch + b) * kHid + h;
    float v = lin[idx] + rr * s;
    s = fmaxf(v, 0.f);
    y[idx] = s;
  }
  hT[b * (2 * kHid) + kHid + h] = s;     // hiddens[:, H:2H]
}

// ---------------------------------------------------------------------------
extern "C" void kernel_launch(void* const* d_in, const int* in_sizes, int n_in,
                              void* d_out, int out_size, void* d_ws, size_t ws_size,
                              hipStream_t stream) {
  const float* x   = (const float*)d_in[0];
  const float* w0  = (const float*)d_in[1];
  const float* b0  = (const float*)d_in[2];
  const float* w1  = (const float*)d_in[3];
  const float* b1  = (const float*)d_in[4];
  const float* rec = (const float*)d_in[5];

  float* out     = (float*)d_out;
  float* hiddens = out + (size_t)kM * kHid;   // +8388608

  // workspace layout (68 MB total)
  char* ws = (char*)d_ws;
  float*  lin = (float*)ws;                              // 32 MB fp32
  __bf16* ahi = (__bf16*)(ws + (size_t)32 * (1 << 20));  // 16 MB
  __bf16* alo = (__bf16*)(ws + (size_t)48 * (1 << 20));  // 16 MB
  __bf16* whi = (__bf16*)(ws + (size_t)64 * (1 << 20));  //  2 MB
  __bf16* wlo = (__bf16*)(ws + (size_t)66 * (1 << 20));  //  2 MB

  const int nA = kM * kIn;        // 8388608
  const int nW = kHid * kIn;      // 1048576
  dim3 ggrid(kM / 128, kHid / 128);  // 64 x 8 blocks
  dim3 gblk(128);
  const int scanBlocks = (kBatch * kHid) / 256;

  // ---- layer 0 ----
  split_bf16_kernel<<<(nA + 255) / 256, 256, 0, stream>>>(x, ahi, alo, nA);
  split_bf16_kernel<<<(nW + 255) / 256, 256, 0, stream>>>(w0, whi, wlo, nW);
  gemm_bf16x3_kernel<<<ggrid, gblk, 0, stream>>>(ahi, alo, whi, wlo, b0, lin);
  indrnn_scan0_kernel<<<scanBlocks, 256, 0, stream>>>(lin, rec, ahi, alo, hiddens);

  // ---- layer 1 (y0 already in ahi/alo as bf16 hi/lo) ----
  split_bf16_kernel<<<(nW + 255) / 256, 256, 0, stream>>>(w1, whi, wlo, nW);
  gemm_bf16x3_kernel<<<ggrid, gblk, 0, stream>>>(ahi, alo, whi, wlo, b1, lin);
  indrnn_scan1_kernel<<<scanBlocks, 256, 0, stream>>>(lin, rec + kHid, out, hiddens);
}
